// unit_gcn_84670985273835
// MI455X (gfx1250) — matrix operand order repaired and checked by reading, not compiled, not run on hardware
//
#include <hip/hip_runtime.h>

// ---------------------------------------------------------------------------
// Fused unit-GCN for MI455X (gfx1250):
//   1x1 conv (v_wmma_f32_16x16x32_bf16) -> graph aggregation (WMMA bf16)
//   -> BatchNorm(eval) + identity residual + ReLU
// Shapes: x (64,64,256,25) f32, A/PA (3,25,25), Wc (192,64), bc (192),
//         gamma/beta/mean/var (64). Output (64,64,256,25) f32.
// One workgroup (8 wave32) handles one (n, 8-row t-tile). grid = 64*32.
// x tile is staged to LDS via GLOBAL_LOAD_ASYNC_TO_LDS_B128 (ASYNCcnt,
// inline asm -- portable across toolchains), then converted in-LDS to bf16.
// ---------------------------------------------------------------------------

typedef __attribute__((ext_vector_type(16))) __bf16 v16bf;
typedef __attribute__((ext_vector_type(8)))  float  v8f;

union FragU { unsigned int u[8]; v16bf v; };

// Async 16B global->LDS copy (per-lane addresses). vdst = LDS byte address
// (low 32 bits of the generic pointer), vaddr = 64-bit global address.
__device__ __forceinline__ void async_copy16(void* lds_dst, const void* gsrc) {
    const unsigned lds_off = (unsigned)(unsigned long long)lds_dst;
    asm volatile("global_load_async_to_lds_b128 %0, %1, off"
                 :: "v"(lds_off), "v"(gsrc) : "memory");
}

__device__ __forceinline__ void async_wait0() {
#if defined(__has_builtin) && __has_builtin(__builtin_amdgcn_s_wait_asynccnt)
    __builtin_amdgcn_s_wait_asynccnt(0);
#else
    asm volatile("s_wait_asynccnt 0" ::: "memory");
#endif
}

// A-matrix fragment (16x32 bf16, M x K), LDS row-major [row][ld].
// ISA layout: lane L -> m = L%16; VGPR r holds K-pair at
//   k = (r<4 ? 2r : 16+2(r-4)) + 8*(L/16)
__device__ __forceinline__ v16bf frag_a(const __bf16* lds,
                                        int row0, int k0, int ld, int lane) {
    const int m = lane & 15, half = lane >> 4;
    const __bf16* base = lds + (row0 + m) * ld + k0 + 8 * half;
    FragU f;
#pragma unroll
    for (int r = 0; r < 8; ++r) {
        const int k = (r < 4) ? (2 * r) : (8 + 2 * r);     // == 16 + 2*(r-4)
        f.u[r] = *(const unsigned int*)(base + k);
    }
    return f.v;
}

// B-matrix fragment (32x16 bf16, K x N), LDS stored TRANSPOSED [col][ld] so
// each lane reads contiguous K-pairs.
// ISA layout: lane L -> n = L%16; VGPR r holds K-pair at
//   k = (r<4 ? 2r : 16+2(r-4)) + 8*(L/16)
__device__ __forceinline__ v16bf frag_b(const __bf16* ldsT,
                                        int col0, int k0, int ld, int lane) {
    const int n = lane & 15, half = lane >> 4;
    const __bf16* base = ldsT + (col0 + n) * ld + k0 + 8 * half;
    FragU f;
#pragma unroll
    for (int r = 0; r < 8; ++r) f.u[r] = *(const unsigned int*)(base + ((r < 4) ? (2 * r) : (8 + 2 * r)));
    return f.v;
}

#define TT    8      // t rows per workgroup
#define COLS  200    // TT*V
#define COLP  208    // COLS padded to 16
#define KVP   96     // K*V=75 padded to 32-multiple
#define ROWS  512    // Cout*TT

__global__ __launch_bounds__(256)
void unit_gcn_fused_kernel(const float* __restrict__ x,
                           const float* __restrict__ A,
                           const float* __restrict__ PA,
                           const float* __restrict__ Wc,
                           const float* __restrict__ bc,
                           const float* __restrict__ gamma,
                           const float* __restrict__ beta,
                           const float* __restrict__ mean,
                           const float* __restrict__ var,
                           float* __restrict__ out) {
    // big-LDS (CDNA5: 320KB/WGP). ~157KB total.
    __shared__ __align__(16) __bf16 xs [COLP * 64];          // x tile,  [col][c]   (B of step A)
    __shared__ __align__(16) __bf16 Wcs[192 * 64];           // weights, [o][c]     (A of step A)
    __shared__ __align__(16) unsigned char y2raw[ROWS * KVP * 2]; // y2 / raw-x staging overlay
    __shared__ __align__(16) __bf16 Abt[32 * KVP];           // Aeff^T,  [w][k*25+v] (B of step B)
    __shared__ float bcs[192];
    __shared__ float scs[64], shs[64];

    __bf16* y2   = (__bf16*)y2raw;   // [c*8+t][k*25+v] (A of step B), phase >= 2
    float*  xraw = (float*)y2raw;    // [c][200] raw fp32 x tile, phase 1 only

    const int tid  = threadIdx.x;
    const int lane = tid & 31;
    const int wv   = tid >> 5;
    const int nb   = blockIdx.x >> 5;        // batch index 0..63
    const int t0   = (blockIdx.x & 31) * TT; // t tile origin

    // ---------------- Phase 1: stage operands into LDS ----------------
    // async-DMA the raw fp32 x tile (64 rows x 200 floats, rows are 800B-aligned
    // and contiguous in (t,v)) into the y2 overlay, 16B per lane per issue.
    for (int idx = tid; idx < 64 * 50; idx += 256) {
        const int c = idx / 50, q = idx % 50;
        const float* g = x + ((nb * 64 + c) * 256 + t0) * 25 + q * 4;
        async_copy16(xraw + c * 200 + q * 4, g);
    }
    // conv weights -> bf16 (hardware cvt)
    for (int idx = tid; idx < 192 * 64; idx += 256) Wcs[idx] = (__bf16)Wc[idx];
    // Aeff = A*PA -> bf16, transposed [w][kv], padded
    for (int idx = tid; idx < 32 * KVP; idx += 256) {
        const int w = idx / KVP, kv = idx % KVP;
        float val = 0.0f;
        if (w < 25 && kv < 75) {
            const int k = kv / 25, v = kv - k * 25;
            const int ai = (k * 25 + v) * 25 + w;
            val = A[ai] * PA[ai];
        }
        Abt[idx] = (__bf16)val;
    }
    // bias + fused BN affine
    if (tid < 192) bcs[tid] = bc[tid];
    if (tid < 64) {
        const float inv = rsqrtf(var[tid] + 1e-5f);
        const float sc  = gamma[tid] * inv;
        scs[tid] = sc;
        shs[tid] = beta[tid] - mean[tid] * sc;
    }

    async_wait0();
    __syncthreads();

    // convert raw fp32 tile (LDS) -> bf16, transposed [col][c]; pad cols 200..207
    for (int idx = tid; idx < COLP * 64; idx += 256) {
        const int c = idx / COLP, col = idx % COLP;
        const float val = (col < COLS) ? xraw[c * 200 + col] : 0.0f;
        xs[col * 64 + c] = (__bf16)val;
    }
    __syncthreads();

    // y2 overlay is now dead as xraw: zero its K padding (kv = 75..95).
    // Disjoint from phase-2's y2 writes, so no extra barrier needed.
    for (int idx = tid; idx < ROWS * (KVP - 75); idx += 256) {
        const int row = idx / (KVP - 75), kv = 75 + idx % (KVP - 75);
        y2[row * KVP + kv] = (__bf16)0.0f;
    }

    // ---------------- Phase 2: y = Wc @ x  (192 x 208, K=64) ----------------
    // 12 M-tiles x 13 N-tiles, 2 K-steps each.
    for (int tile = wv; tile < 12 * 13; tile += 8) {
        const int mt = tile % 12, nt = tile / 12;
        v8f acc = {};
#pragma unroll
        for (int ks = 0; ks < 2; ++ks) {
            v16bf a = frag_a(Wcs, mt * 16, ks * 32, 64, lane);
            v16bf b = frag_b(xs,  nt * 16, ks * 32, 64, lane);
            acc = __builtin_amdgcn_wmma_f32_16x16x32_bf16(
                      false, a, false, b, (short)0, acc, false, false);
        }
        // scatter-store y+bc into step-B layout: row=(c*8+t), col=(k*25+v)
        const int nl = lane & 15, half = lane >> 4;
        const int colg = nt * 16 + nl;
        if (colg < COLS) {
            const int t = colg / 25, v = colg - t * 25;
#pragma unroll
            for (int r = 0; r < 8; ++r) {
                const int o = mt * 16 + r + 8 * half;      // 0..191
                const int c = o & 63, k = o >> 6;
                y2[(c * TT + t) * KVP + k * 25 + v] = (__bf16)(acc[r] + bcs[o]);
            }
        }
    }
    __syncthreads();

    // ------- Phase 3: z = y2 @ Aeff (512 x 32, K=96) + BN + res + ReLU -------
    // 32 M-tiles x 2 N-tiles, 3 K-steps each.
    for (int tile = wv; tile < 64; tile += 8) {
        const int mt = tile >> 1, nt = tile & 1;
        v8f acc = {};
#pragma unroll
        for (int ks = 0; ks < 3; ++ks) {
            v16bf a = frag_a(y2,  mt * 16, ks * 32, KVP, lane);
            v16bf b = frag_b(Abt, nt * 16, ks * 32, KVP, lane);
            acc = __builtin_amdgcn_wmma_f32_16x16x32_bf16(
                      false, a, false, b, (short)0, acc, false, false);
        }
        const int nl = lane & 15, half = lane >> 4;
        const int w = nt * 16 + nl;
        if (w < 25) {
#pragma unroll
            for (int r = 0; r < 8; ++r) {
                const int row = mt * 16 + r + 8 * half;    // c*8 + t
                const int c = row >> 3, t = row & 7;
                const int gi = ((nb * 64 + c) * 256 + (t0 + t)) * 25 + w;
                const float val = acc[r] * scs[c] + shs[c] + x[gi]; // BN + residual
                out[gi] = fmaxf(val, 0.0f);
            }
        }
    }
}

extern "C" void kernel_launch(void* const* d_in, const int* in_sizes, int n_in,
                              void* d_out, int out_size, void* d_ws, size_t ws_size,
                              hipStream_t stream) {
    (void)in_sizes; (void)n_in; (void)out_size; (void)d_ws; (void)ws_size;
    const float* x     = (const float*)d_in[0];
    const float* A     = (const float*)d_in[1];
    const float* PA    = (const float*)d_in[2];
    const float* Wc    = (const float*)d_in[3];
    const float* bc    = (const float*)d_in[4];
    const float* gamma = (const float*)d_in[5];
    const float* beta  = (const float*)d_in[6];
    const float* mean  = (const float*)d_in[7];
    const float* var   = (const float*)d_in[8];
    float* out = (float*)d_out;

    dim3 grid(64 * 32);   // (n, t-tile)
    dim3 block(256);      // 8 wave32
    unit_gcn_fused_kernel<<<grid, block, 0, stream>>>(
        x, A, PA, Wc, bc, gamma, beta, mean, var, out);
}
